// RNNBlock_58858231824557
// MI455X (gfx1250) — compile-verified, matrix operand order checked by reading, not run on hardware
//
#include <hip/hip_runtime.h>

// ---------------------------------------------------------------------------
// 2-layer LSTM (B=64, T=1024, D=H=512) for gfx1250 (MI455X).
//   Phase A: convert X / W to bf16 (W transposed to N-major for fragment loads)
//   Phase B: big WMMA GEMM  Zx = X @ W_x  (all timesteps at once)
//            -> LDS-tiled, double-buffered with global_load_async_to_lds_b128
//   Phase C: per-timestep fused kernel: zh = h_{t-1} @ W_h (WMMA bf16, A tile
//            staged in LDS via async copy), z = Zx + zh + b, gates in f32.
// ---------------------------------------------------------------------------

typedef __attribute__((ext_vector_type(16))) __bf16 bf16x16;
typedef __attribute__((ext_vector_type(8)))  __bf16 bf16x8;
typedef __attribute__((ext_vector_type(8)))  float  f32x8;

#define B_BATCH 64
#define T_SEQ   1024
#define D_IN    512
#define H_DIM   512
#define KTOT    1024                 // D + H (rows of W)
#define N4H     2048                 // 4H   (cols of W)
#define M_ROWS  (B_BATCH * T_SEQ)    // 65536

__device__ __forceinline__ unsigned short f32_to_bf16_rne(float f) {
    unsigned int u = __float_as_uint(f);
    u += 0x7FFFu + ((u >> 16) & 1u);   // round-to-nearest-even
    return (unsigned short)(u >> 16);
}

__device__ __forceinline__ float sigmoid_fast(float x) {
    return 1.0f / (1.0f + __expf(-x));
}
__device__ __forceinline__ float tanh_fast(float x) {
    float e = __expf(-2.0f * x);
    return (1.0f - e) / (1.0f + e);
}

// --------------------- CDNA5 async global->LDS copy ------------------------
// LDS aperture occupies addr[63:32]; low 32 bits of a generic LDS pointer are
// the LDS byte offset (ISA 10.2 aperture check), usable as the VDST operand.
__device__ __forceinline__ unsigned lds_off(const void* p) {
    return (unsigned)(unsigned long long)p;
}
// One 16-byte async copy per lane: LDS[vdst] = MEM[vaddr..vaddr+15]; ASYNCcnt.
__device__ __forceinline__ void async_copy_b128(unsigned lds_addr, const void* gaddr) {
    asm volatile("global_load_async_to_lds_b128 %0, %1, off"
                 :: "v"(lds_addr), "v"(gaddr) : "memory");
}
__device__ __forceinline__ void wait_async_all() {
    asm volatile("s_wait_asynccnt 0x0" ::: "memory");
}

// Load one 16x32 bf16 A/B fragment for this lane: two contiguous 16B quads
// at K-offset +0 and +16 elements (caller bakes laneHalf*8 into p).
// Works for global (global_load_b128) and LDS (ds_load_b128) pointers.
__device__ __forceinline__ bf16x16 load_ab_frag(const unsigned short* p) {
    bf16x8 lo = *reinterpret_cast<const bf16x8*>(p);
    bf16x8 hi = *reinterpret_cast<const bf16x8*>(p + 16);
    union { bf16x16 v; bf16x8 h[2]; } u;
    u.h[0] = lo; u.h[1] = hi;
    return u.v;
}

#define WMMA_BF16(A, Bm, C) \
    __builtin_amdgcn_wmma_f32_16x16x32_bf16(false, (A), false, (Bm), (short)0, (C), false, false)

// ------------------------------ conversions --------------------------------

__global__ void __launch_bounds__(256)
cvt_f32_bf16_kernel(const float* __restrict__ in, unsigned short* __restrict__ out, int n) {
    int i = blockIdx.x * blockDim.x + threadIdx.x;
    if (i < n) out[i] = f32_to_bf16_rne(in[i]);
}

// W [KTOT, N4H] f32 (row-major)  ->  WT [N4H, KTOT] bf16 (N-major)
__global__ void __launch_bounds__(256)
transpose_w_bf16_kernel(const float* __restrict__ W, unsigned short* __restrict__ WT) {
    int idx = blockIdx.x * blockDim.x + threadIdx.x;
    if (idx >= KTOT * N4H) return;
    int n = idx % N4H;
    int k = idx / N4H;                       // coalesced read over n
    WT[(size_t)n * KTOT + k] = f32_to_bf16_rne(W[idx]);
}

__global__ void __launch_bounds__(256)
init_state_kernel(float* __restrict__ c, unsigned short* __restrict__ h0,
                  unsigned short* __restrict__ h1) {
    int i = blockIdx.x * blockDim.x + threadIdx.x;
    if (i < B_BATCH * H_DIM) { c[i] = 0.0f; h0[i] = 0; h1[i] = 0; }
}

// ------------------------- big GEMM: Zx = A @ W_x ---------------------------
// A  : [M_ROWS, 512] bf16 row-major      (X or layer-1 h sequence)
// WT : [2048, 1024]  bf16 N-major, x-part at K offset 0
// Z  : [M_ROWS, 2048] f32
// Block (8 waves): 128x64 C tile. K staged 32 at a time through LDS with
// double-buffered async copies; each wave computes a 16x64 slab.
#define GK        32                         // K per stage
#define A_STG     (128 * GK)                 // 4096 bf16 = 8 KB
#define B_STG     (64 * GK)                  // 2048 bf16 = 4 KB
#define STG_ELEMS (A_STG + B_STG)            // 12 KB / stage

__global__ void __launch_bounds__(256)
gemm_x_bf16_wmma(const unsigned short* __restrict__ A,
                 const unsigned short* __restrict__ WT,
                 float* __restrict__ Z) {
    __shared__ unsigned short smem[2 * STG_ELEMS];   // 24 KB

    const int tid      = threadIdx.x;
    const int wave     = tid >> 5;
    const int lane     = tid & 31;
    const int laneHalf = lane >> 4;
    const int ln16     = lane & 15;
    const int mBlk     = blockIdx.x >> 5;    // 512 blocks of 128 rows
    const int nBlk     = blockIdx.x & 31;    // 32 blocks of 64 cols

    const unsigned short* Ag = A  + (size_t)mBlk * 128 * D_IN;
    const unsigned short* Bg = WT + (size_t)nBlk * 64 * KTOT;
    const unsigned smem_base = lds_off(smem);

    // Issue async copies for one K stage into LDS buffer `stg`.
    auto issue_stage = [&](int stg, int k0) {
        unsigned sb = smem_base + (unsigned)stg * (STG_ELEMS * 2);
        // A tile: 128 rows x 32 K = 512 16B-chunks; 2 per thread.
#pragma unroll
        for (int i = 0; i < 2; ++i) {
            int c   = tid + i * 256;         // 0..511
            int row = c >> 2;                // 4 chunks per row
            int ke  = (c & 3) * 8;           // element offset in K
            async_copy_b128(sb + (unsigned)(row * GK + ke) * 2,
                            Ag + (size_t)row * D_IN + k0 + ke);
        }
        // B tile: 64 n-rows x 32 K = 256 16B-chunks; 1 per thread.
        {
            int c   = tid;                   // 0..255
            int row = c >> 2;
            int ke  = (c & 3) * 8;
            async_copy_b128(sb + (unsigned)(A_STG + row * GK + ke) * 2,
                            Bg + (size_t)row * KTOT + k0 + ke);
        }
    };

    issue_stage(0, 0);

    f32x8 acc0 = {}, acc1 = {}, acc2 = {}, acc3 = {};
    const int nStages = D_IN / GK;           // 16
    for (int s = 0; s < nStages; ++s) {
        wait_async_all();                    // this wave's copies for stage s
        __syncthreads();                     // everyone's copies visible; prior
                                             // readers of the other buffer done
        if (s + 1 < nStages) issue_stage((s + 1) & 1, (s + 1) * GK);

        const unsigned short* sa = smem + (s & 1) * STG_ELEMS;
        const unsigned short* sbp = sa + A_STG;
        const unsigned short* aR = sa  + (wave * 16 + ln16) * GK + laneHalf * 8;
        const unsigned short* bR = sbp + ln16 * GK + laneHalf * 8;
        bf16x16 a  = load_ab_frag(aR);
        bf16x16 b0 = load_ab_frag(bR);
        bf16x16 b1 = load_ab_frag(bR + 16 * GK);
        bf16x16 b2 = load_ab_frag(bR + 32 * GK);
        bf16x16 b3 = load_ab_frag(bR + 48 * GK);
        acc0 = WMMA_BF16(a, b0, acc0);
        acc1 = WMMA_BF16(a, b1, acc1);
        acc2 = WMMA_BF16(a, b2, acc2);
        acc3 = WMMA_BF16(a, b3, acc3);
    }

    const int rowBase = mBlk * 128 + wave * 16 + laneHalf * 8;  // D-matrix rows
#pragma unroll
    for (int r = 0; r < 8; ++r) {
        float* zp = Z + (size_t)(rowBase + r) * N4H + nBlk * 64 + ln16;
        zp[0]  = acc0[r];
        zp[16] = acc1[r];
        zp[32] = acc2[r];
        zp[48] = acc3[r];
    }
}

// -------------------- fused recurrent step (one timestep) -------------------
// hprev/hnext : [64, 512] bf16 (double buffered across t)
// WT          : [2048, 1024] bf16 N-major; recurrent part at K offset 512
// Zx          : [M_ROWS, 2048] f32 precomputed input contribution
// Block = 8 waves sharing one 16-row A tile of h_{t-1} (staged in LDS);
// each wave: one 16x16 tile of H for ALL FOUR gates (A frag reused 4x).
__global__ void __launch_bounds__(256)
lstm_step_wmma(const unsigned short* __restrict__ hprev,
               unsigned short* __restrict__ hnext,
               const unsigned short* __restrict__ WT,
               const float* __restrict__ Zx,
               const float* __restrict__ bias,
               float* __restrict__ c_state,
               unsigned short* __restrict__ h_seq_bf16,  // layer-1 output (or null)
               float* __restrict__ h_seq_f32,            // layer-2 output (or null)
               int t) {
    __shared__ unsigned short As[16 * H_DIM];            // 16 KB shared A tile

    const int tid      = threadIdx.x;
    const int wave     = tid >> 5;
    const int lane     = tid & 31;
    const int laneHalf = lane >> 4;
    const int ln16     = lane & 15;
    const int mTile    = blockIdx.x >> 2;                // same for all 8 waves
    const int nTile    = ((blockIdx.x & 3) << 3) + wave; // 0..31

    // Stage the shared 16x512 bf16 A tile: 1024 16B-chunks, 4 per thread.
    {
        const unsigned as_base = lds_off(As);
        const unsigned short* hg = hprev + (size_t)mTile * 16 * H_DIM;
#pragma unroll
        for (int i = 0; i < 4; ++i) {
            int c = tid + i * 256;                       // 0..1023
            async_copy_b128(as_base + (unsigned)c * 16,
                            hg + (size_t)c * 8);
        }
        wait_async_all();
        __syncthreads();
    }

    const int col = nTile * 16 + ln16;                   // per-gate column
    const size_t gStride = (size_t)H_DIM * KTOT;         // gate-to-gate in WT
    const unsigned short* aRow = As + ln16 * H_DIM + laneHalf * 8;
    const unsigned short* bRow = WT + (size_t)col * KTOT + D_IN + laneHalf * 8;

    f32x8 zi = {}, zj = {}, zf = {}, zo = {};
#pragma unroll
    for (int k0 = 0; k0 < H_DIM; k0 += 32) {
        bf16x16 a   = load_ab_frag(aRow + k0);           // ds_load_b128 x2
        bf16x16 bi_ = load_ab_frag(bRow + k0);
        bf16x16 bj_ = load_ab_frag(bRow + k0 + gStride);
        bf16x16 bf_ = load_ab_frag(bRow + k0 + 2 * gStride);
        bf16x16 bo_ = load_ab_frag(bRow + k0 + 3 * gStride);
        zi = WMMA_BF16(a, bi_, zi);
        zj = WMMA_BF16(a, bj_, zj);
        zf = WMMA_BF16(a, bf_, zf);
        zo = WMMA_BF16(a, bo_, zo);
    }

    const float bi_s = bias[0 * H_DIM + col];
    const float bj_s = bias[1 * H_DIM + col];
    const float bf_s = bias[2 * H_DIM + col] + 1.0f;     // FORGET_BIAS
    const float bo_s = bias[3 * H_DIM + col];

    const int rowBase = mTile * 16 + laneHalf * 8;       // D-matrix layout rows
#pragma unroll
    for (int r = 0; r < 8; ++r) {
        const int b = rowBase + r;                       // batch index
        const size_t zrow = ((size_t)b * T_SEQ + t) * N4H;
        float vi = zi[r] + Zx[zrow + 0 * H_DIM + col] + bi_s;
        float vj = zj[r] + Zx[zrow + 1 * H_DIM + col] + bj_s;
        float vf = zf[r] + Zx[zrow + 2 * H_DIM + col] + bf_s;
        float vo = zo[r] + Zx[zrow + 3 * H_DIM + col] + bo_s;

        const int ci = b * H_DIM + col;
        float c = c_state[ci];
        c = c * sigmoid_fast(vf) + sigmoid_fast(vi) * tanh_fast(vj);
        float h = tanh_fast(c) * sigmoid_fast(vo);
        c_state[ci] = c;

        unsigned short hb = f32_to_bf16_rne(h);
        hnext[ci] = hb;

        const size_t orow = ((size_t)b * T_SEQ + t) * H_DIM + col;
        if (h_seq_bf16) h_seq_bf16[orow] = hb;
        if (h_seq_f32)  h_seq_f32[orow]  = h;
    }
}

// ------------------------------ host driver --------------------------------

extern "C" void kernel_launch(void* const* d_in, const int* in_sizes, int n_in,
                              void* d_out, int out_size, void* d_ws, size_t ws_size,
                              hipStream_t stream) {
    (void)in_sizes; (void)n_in; (void)out_size; (void)ws_size;
    const float* X  = (const float*)d_in[0];
    const float* W1 = (const float*)d_in[1];
    const float* b1 = (const float*)d_in[2];
    const float* W2 = (const float*)d_in[3];
    const float* b2 = (const float*)d_in[4];
    float* out = (float*)d_out;

    // Workspace carve-up (all offsets are multiples of 64KB).
    char* p = (char*)d_ws;
    unsigned short* Xbf  = (unsigned short*)p; p += (size_t)M_ROWS * D_IN * 2;   //  64 MB
    unsigned short* H1   = (unsigned short*)p; p += (size_t)M_ROWS * H_DIM * 2;  //  64 MB
    unsigned short* W1T  = (unsigned short*)p; p += (size_t)N4H * KTOT * 2;      //   4 MB
    unsigned short* W2T  = (unsigned short*)p; p += (size_t)N4H * KTOT * 2;      //   4 MB
    float*          Zx   = (float*)p;          p += (size_t)M_ROWS * N4H * 4;    // 512 MB
    float*          Cst  = (float*)p;          p += (size_t)B_BATCH * H_DIM * 4; // 128 KB
    unsigned short* Hst0 = (unsigned short*)p; p += (size_t)B_BATCH * H_DIM * 2; //  64 KB
    unsigned short* Hst1 = (unsigned short*)p;                                   //  64 KB

    // Phase A: precision conversion + weight transpose (one-time cost).
    {
        int n = M_ROWS * D_IN;
        cvt_f32_bf16_kernel<<<(n + 255) / 256, 256, 0, stream>>>(X, Xbf, n);
        int nw = KTOT * N4H;
        transpose_w_bf16_kernel<<<(nw + 255) / 256, 256, 0, stream>>>(W1, W1T);
        transpose_w_bf16_kernel<<<(nw + 255) / 256, 256, 0, stream>>>(W2, W2T);
    }

    const int gemmBlocks = (M_ROWS / 128) * (N4H / 64);  // 512 * 32 = 16384
    const int stBlocks   = (B_BATCH * H_DIM + 255) / 256;

    // ---------------- Layer 1 ----------------
    init_state_kernel<<<stBlocks, 256, 0, stream>>>(Cst, Hst0, Hst1);
    gemm_x_bf16_wmma<<<gemmBlocks, 256, 0, stream>>>(Xbf, W1T, Zx);
    for (int t = 0; t < T_SEQ; ++t) {
        const unsigned short* hp = (t & 1) ? Hst1 : Hst0;
        unsigned short*       hn = (t & 1) ? Hst0 : Hst1;
        lstm_step_wmma<<<16, 256, 0, stream>>>(hp, hn, W1T, Zx, b1, Cst,
                                               H1, nullptr, t);
    }

    // ---------------- Layer 2 ----------------
    init_state_kernel<<<stBlocks, 256, 0, stream>>>(Cst, Hst0, Hst1);
    gemm_x_bf16_wmma<<<gemmBlocks, 256, 0, stream>>>(H1, W2T, Zx);
    for (int t = 0; t < T_SEQ; ++t) {
        const unsigned short* hp = (t & 1) ? Hst1 : Hst0;
        unsigned short*       hn = (t & 1) ? Hst0 : Hst1;
        lstm_step_wmma<<<16, 256, 0, stream>>>(hp, hn, W2T, Zx, b2, Cst,
                                               nullptr, out, t);
    }
}
